// NONLocalNet_27882927686219
// MI455X (gfx1250) — matrix operand order repaired and checked by reading, not compiled, not run on hardware
//
#include <hip/hip_runtime.h>

typedef __attribute__((ext_vector_type(16))) _Float16 v16h;
typedef __attribute__((ext_vector_type(8)))  float    v8f;
typedef __attribute__((ext_vector_type(4)))  unsigned int v4u;
typedef __attribute__((ext_vector_type(4)))  unsigned int u32x4;
typedef __attribute__((ext_vector_type(8)))  int          i32x8;
typedef __attribute__((ext_vector_type(4)))  int          i32x4;

// B=256, C=2, H=64, W=16 -> N=1024 positions, Np=256 pooled, IC=64
#define NB   256
#define NPOS 1024
#define NPL  256
#define IC   64

union FragH { v16h h; v4u u[2]; };

__device__ __forceinline__ v8f wmma_f16(v16h a, v16h b, v8f c) {
    // D = A(16x32 f16) * B(32x16 f16) + C(16x16 f32)
    return __builtin_amdgcn_wmma_f32_16x16x32_f16(
        /*neg_a=*/false, a, /*neg_b=*/false, b,
        /*c_mod=*/(short)0, c, /*reuse_a=*/false, /*reuse_b=*/false);
}

#if defined(__has_builtin)
#if __has_builtin(__builtin_amdgcn_tensor_load_to_lds) && __has_builtin(__builtin_amdgcn_s_wait_tensorcnt)
#define USE_TDM 1
#endif
#endif
#ifndef USE_TDM
#define USE_TDM 0
#endif

#if USE_TDM
// TDM: contiguous 32 KB (16384 x 2B elements) global -> LDS copy.
// D# per CDNA5 ISA:
//   group0: count=1 [1:0], lds_addr [63:32], global_addr [120:64], type=2 [127:126]
//   group1: wg_mask=0 [15:0], data_size=1 (2B) [17:16],
//           tensor_dim0=16384 [79:48], tensor_dim1=1 [111:80],
//           tile_dim0=16384 [127:112], tile_dim1/2=0,
//           tensor_dim0_stride=16384 [207:160]
__device__ __forceinline__ void tdm_load_32k(const void* gsrc, unsigned ldsOff) {
    unsigned long long ga = (unsigned long long)(uintptr_t)gsrc;
    u32x4 g0;
    g0[0] = 1u;                                   // count=1
    g0[1] = ldsOff;                               // lds_addr
    g0[2] = (unsigned)(ga & 0xFFFFFFFFu);         // global_addr[31:0]
    g0[3] = (unsigned)(ga >> 32) | (2u << 30);    // global_addr[56:32] | type=2
    i32x8 g1;
    g1[0] = 0x00010000;                           // data_size=1 (2 bytes)
    g1[1] = 0x40000000;                           // tensor_dim0[15:0]=16384 at [31:16]
    g1[2] = 0x00010000;                           // tensor_dim1[15:0]=1 at [31:16]
    g1[3] = 0x40000000;                           // tile_dim0=16384 at [31:16]
    g1[4] = 0;                                    // tile_dim1=0, tile_dim2=0
    g1[5] = 16384;                                // tensor_dim0_stride[31:0]
    g1[6] = 0;
    g1[7] = 0;
    i32x4 gz = {0, 0, 0, 0};
#if defined(__clang_major__) && (__clang_major__ >= 23)
    i32x8 gz8 = {0, 0, 0, 0, 0, 0, 0, 0};
    __builtin_amdgcn_tensor_load_to_lds(g0, g1, gz, gz, gz8, 0);
#else
    __builtin_amdgcn_tensor_load_to_lds(g0, g1, gz, gz, 0);
#endif
}
#endif

// ---------------- projection kernels ----------------

// theta[b][n][o] = Wth[o,0]*y[b,0,n] + Wth[o,1]*y[b,1,n] + bth[o]   (f16 out)
__global__ void theta_kernel(const float* __restrict__ y,
                             const float* __restrict__ Wth,
                             const float* __restrict__ bth,
                             _Float16* __restrict__ theta)
{
    int e = blockIdx.x * blockDim.x + threadIdx.x;   // B*N*IC threads
    int o = e & 63;
    int n = (e >> 6) & 1023;
    int b = e >> 16;
    float y0 = y[b * 2048 + n];
    float y1 = y[b * 2048 + 1024 + n];
    float v  = Wth[o * 2] * y0 + Wth[o * 2 + 1] * y1 + bth[o];
    theta[e] = (_Float16)v;
}

// pooled projections: phiT[b][m][o], gT[b][o][m]  (2x2 maxpool of 1x1 conv)
__global__ void pool_proj_kernel(const float* __restrict__ x,
                                 const float* __restrict__ Wg,  const float* __restrict__ bg,
                                 const float* __restrict__ Wphi,const float* __restrict__ bphi,
                                 _Float16* __restrict__ phiT, _Float16* __restrict__ gT)
{
    int e = blockIdx.x * blockDim.x + threadIdx.x;   // B*IC*Np threads
    int m = e & 255;
    int o = (e >> 8) & 63;
    int b = e >> 14;
    int ph = m >> 3, pw = m & 7;
    const float* x0 = x + (size_t)b * 2048;
    const float* x1 = x0 + 1024;
    float g0 = Wg[o * 2],   g1 = Wg[o * 2 + 1];
    float p0 = Wphi[o * 2], p1 = Wphi[o * 2 + 1];
    float gv = -3.4e38f, pv = -3.4e38f;
#pragma unroll
    for (int dh = 0; dh < 2; ++dh)
#pragma unroll
        for (int dw = 0; dw < 2; ++dw) {
            int n = (ph * 2 + dh) * 16 + (pw * 2 + dw);
            float a = x0[n], c = x1[n];
            gv = fmaxf(gv, g0 * a + g1 * c);
            pv = fmaxf(pv, p0 * a + p1 * c);
        }
    gv += bg[o];
    pv += bphi[o];
    gT[(size_t)b * 16384 + o * 256 + m]  = (_Float16)gv;
    phiT[(size_t)b * 16384 + m * 64 + o] = (_Float16)pv;
}

__global__ void init_out(float* __restrict__ out, const float* __restrict__ bfc, int n)
{
    int i = blockIdx.x * blockDim.x + threadIdx.x;
    if (i < n) out[i] = bfc[i & 1];
}

// ---------------- fused attention kernel ----------------
// one block per batch; 8 waves; each wave handles 8 query tiles of 16 rows.
// LDS map (dynamic, 256 KB):
//   [0,32768)        phiT tile  (256x64 f16)
//   [32768,65536)    gT tile    (64x256 f16)
//   [65536,196608)   per-wave scores, 16x256 f32 (16 KB each)
//   [196608,262144)  per-wave P,      16x256 f16 ( 8 KB each)
__global__ __launch_bounds__(256) void attn_kernel(
    const _Float16* __restrict__ theta,  // (B,1024,64)
    const _Float16* __restrict__ phiT,   // (B,256,64)
    const _Float16* __restrict__ gT,     // (B,64,256)
    const float* __restrict__ Ww,  const float* __restrict__ bw,
    const float* __restrict__ bn_w, const float* __restrict__ bn_b,
    const float* __restrict__ bn_mean, const float* __restrict__ bn_var,
    const float* __restrict__ Wfc,
    float* __restrict__ out)
{
    extern __shared__ char smem[];
    _Float16* phiL = (_Float16*)smem;
    _Float16* gL   = (_Float16*)(smem + 32768);

    int b    = blockIdx.x;
    int tid  = threadIdx.x;
    int wave = tid >> 5;
    int lane = tid & 31;
    int row16 = lane & 15;
    bool hi  = lane >= 16;
    int aoff = hi ? 8 : 0;    // A-fragment lane K offset (halfs)
    int boff = hi ? 8 : 0;    // B-fragment lane K offset (halfs)

    float*    scoresW = (float*)(smem + 65536) + wave * 4096;     // 16x256 f32
    _Float16* Pw      = (_Float16*)(smem + 196608) + wave * 4096; // 16x256 f16

    // stage phiT / gT tiles for this batch
#if USE_TDM
    if (wave == 0) {
        tdm_load_32k(phiT + (size_t)b * 16384, 0u);
        tdm_load_32k(gT   + (size_t)b * 16384, 32768u);
        __builtin_amdgcn_s_wait_tensorcnt(0);
    }
#else
    {
        const v4u* s0 = (const v4u*)(phiT + (size_t)b * 16384);
        const v4u* s1 = (const v4u*)(gT   + (size_t)b * 16384);
        v4u* d0 = (v4u*)phiL;
        v4u* d1 = (v4u*)gL;
        for (int i = tid; i < 2048; i += 256) d0[i] = s0[i];
        for (int i = tid; i < 2048; i += 256) d1[i] = s1[i];
    }
#endif
    __syncthreads();

    for (int rt = wave; rt < 64; rt += 8) {
        // ---- A fragments: theta rows rt*16..rt*16+15, K = 0..63 (two k-steps)
        const _Float16* tb = theta + (size_t)b * 65536 + (size_t)(rt * 16 + row16) * 64;
        __builtin_prefetch(tb + 8 * 16 * 64, 0, 0);   // next tile for this wave
        FragH a0, a1;
        a0.u[0] = *(const v4u*)(tb + aoff);
        a0.u[1] = *(const v4u*)(tb + aoff + 16);
        a1.u[0] = *(const v4u*)(tb + 32 + aoff);
        a1.u[1] = *(const v4u*)(tb + 32 + aoff + 16);

        // ---- scores f = theta * phi  (16 x 256)
        for (int nt = 0; nt < 16; ++nt) {
            const _Float16* pb = phiL + (nt * 16 + row16) * 64;
            FragH b0, b1;
            b0.u[0] = *(const v4u*)(pb + boff);
            b0.u[1] = *(const v4u*)(pb + boff + 16);
            b1.u[0] = *(const v4u*)(pb + 32 + boff);
            b1.u[1] = *(const v4u*)(pb + 32 + boff + 16);
            v8f acc = {};
            acc = wmma_f16(a0.h, b0.h, acc);
            acc = wmma_f16(a1.h, b1.h, acc);
            float* srow = scoresW + (hi ? 8 * 256 : 0) + nt * 16 + row16;
#pragma unroll
            for (int r = 0; r < 8; ++r) srow[r * 256] = acc[r];
        }

        // ---- softmax over each of the 16 rows (all 32 lanes cooperate)
        for (int r = 0; r < 16; ++r) {
            const float* rowp = scoresW + r * 256;
            float v[8];
            float mx = -3.4e38f;
#pragma unroll
            for (int i = 0; i < 8; ++i) { v[i] = rowp[lane + 32 * i]; mx = fmaxf(mx, v[i]); }
#pragma unroll
            for (int s = 1; s < 32; s <<= 1) mx = fmaxf(mx, __shfl_xor(mx, s, 32));
            float sum = 0.f;
#pragma unroll
            for (int i = 0; i < 8; ++i) { v[i] = __expf(v[i] - mx); sum += v[i]; }
#pragma unroll
            for (int s = 1; s < 32; s <<= 1) sum += __shfl_xor(sum, s, 32);
            float rs = 1.0f / sum;
            _Float16* prow = Pw + r * 256;
#pragma unroll
            for (int i = 0; i < 8; ++i) prow[lane + 32 * i] = (_Float16)(v[i] * rs);
        }

        // ---- z = P (16x256) * g (256x64)
        v8f z[4] = {};
        for (int ks = 0; ks < 8; ++ks) {
            int kb = ks * 32;
            const _Float16* pr = Pw + row16 * 256 + kb + aoff;
            FragH af;
            af.u[0] = *(const v4u*)(pr);
            af.u[1] = *(const v4u*)(pr + 16);
#pragma unroll
            for (int nt = 0; nt < 4; ++nt) {
                const _Float16* gb = gL + (nt * 16 + row16) * 256 + kb + boff;
                FragH bf;
                bf.u[0] = *(const v4u*)(gb);
                bf.u[1] = *(const v4u*)(gb + 16);
                z[nt] = wmma_f16(af.h, bf.h, z[nt]);
            }
        }

        // ---- stash z tile (16 rows x 64 ch) in LDS (reuse scores region)
        float* zl = scoresW;
#pragma unroll
        for (int nt = 0; nt < 4; ++nt)
#pragma unroll
            for (int r = 0; r < 8; ++r)
                zl[(r + (hi ? 8 : 0)) * 64 + nt * 16 + row16] = z[nt][r];

        // ---- epilogue: conv back to C=2, BN, FC contribution
        // lanes 0..15 -> o=0, lanes 16..31 -> o=1; each lane one row
        int row = row16;
        int o = hi ? 1 : 0;
        float wz = bw[o];
        for (int c = 0; c < 64; ++c) wz += zl[row * 64 + c] * Ww[o * 64 + c];
        float inv = rsqrtf(bn_var[o] + 1e-5f);
        float val = (wz - bn_mean[o]) * inv * bn_w[o] + bn_b[o];
        int pos = rt * 16 + row;                       // flat index = o*1024 + pos
        float c0 = val * Wfc[o * 1024 + pos];
        float c1 = val * Wfc[2048 + o * 1024 + pos];
#pragma unroll
        for (int s = 1; s < 32; s <<= 1) {
            c0 += __shfl_xor(c0, s, 32);
            c1 += __shfl_xor(c1, s, 32);
        }
        if (lane == 0) {
            atomicAdd(&out[b * 2 + 0], c0);
            atomicAdd(&out[b * 2 + 1], c1);
        }
    }
}

// ---------------- launch ----------------

extern "C" void kernel_launch(void* const* d_in, const int* in_sizes, int n_in,
                              void* d_out, int out_size, void* d_ws, size_t ws_size,
                              hipStream_t stream)
{
    const float* y       = (const float*)d_in[0];
    const float* x       = (const float*)d_in[1];
    const float* Wg      = (const float*)d_in[2];
    const float* bg      = (const float*)d_in[3];
    const float* Wth     = (const float*)d_in[4];
    const float* bth     = (const float*)d_in[5];
    const float* Wphi    = (const float*)d_in[6];
    const float* bphi    = (const float*)d_in[7];
    const float* Ww      = (const float*)d_in[8];
    const float* bw      = (const float*)d_in[9];
    const float* bn_w    = (const float*)d_in[10];
    const float* bn_b    = (const float*)d_in[11];
    const float* bn_mean = (const float*)d_in[12];
    const float* bn_var  = (const float*)d_in[13];
    const float* Wfc     = (const float*)d_in[14];
    const float* bfc     = (const float*)d_in[15];
    float* out = (float*)d_out;

    char* ws = (char*)d_ws;
    _Float16* theta = (_Float16*)ws;                         // 256*1024*64 f16 = 32 MB
    _Float16* phiT  = (_Float16*)(ws + 33554432);            // 256*256*64 f16 =  8 MB
    _Float16* gT    = (_Float16*)(ws + 33554432 + 8388608);  // 256*64*256 f16 =  8 MB

    init_out<<<2, 256, 0, stream>>>(out, bfc, out_size);
    theta_kernel<<<(NB * NPOS * IC) / 256, 256, 0, stream>>>(y, Wth, bth, theta);
    pool_proj_kernel<<<(NB * IC * NPL) / 256, 256, 0, stream>>>(x, Wg, bg, Wphi, bphi, phiT, gT);
    attn_kernel<<<NB, 256, 262144, stream>>>(theta, phiT, gT, Ww, bw,
                                             bn_w, bn_b, bn_mean, bn_var, Wfc, out);
}
